// SocialGraphEncoder_14972255994488
// MI455X (gfx1250) — compile-verified, matrix operand order checked by reading, not run on hardware
//
#include <hip/hip_runtime.h>
#include <hip/hip_bf16.h>
#include <math.h>

#define N_NODES 100000
#define N_EDGES 1600000
#define ETOT    (N_EDGES + N_NODES)
#define DIN  16
#define H    128
#define HEADS 4
#define DOUT 32
#define NGRAPH 64
#define FOUT 256
#define NEG_SLOPE 0.2f

typedef __attribute__((ext_vector_type(16))) _Float16 v16h;
typedef __attribute__((ext_vector_type(8)))  float    v8f;

// ---------------------------------------------------------------- helpers
__device__ __forceinline__ float gelu_exact(float x) {
    return 0.5f * x * (1.0f + erff(x * 0.70710678118654752f));
}
// order-preserving float<->uint encoding for atomicMax on floats
__device__ __forceinline__ unsigned fenc(float f) {
    unsigned b = __float_as_uint(f);
    return (b & 0x80000000u) ? ~b : (b | 0x80000000u);
}
__device__ __forceinline__ float fdec(unsigned k) {
    return (k & 0x80000000u) ? __uint_as_float(k & 0x7FFFFFFFu)
                             : __uint_as_float(~k);
}
__device__ __forceinline__ void edge_endpoints(const int* __restrict__ ei, int e,
                                               int& s, int& d) {
    if (e < N_EDGES) { s = ei[e]; d = ei[N_EDGES + e]; }
    else             { s = e - N_EDGES; d = s; }   // self loops
}

// ------------------------------------------------ 1) input projection + GELU
__global__ void proj_gelu_kernel(const float* __restrict__ x,
                                 const float* __restrict__ Wp,
                                 const float* __restrict__ bp,
                                 float* __restrict__ h) {
    int tid = blockIdx.x * blockDim.x + threadIdx.x;
    if (tid >= N_NODES * H) return;
    int n = tid >> 7, c = tid & 127;
    float acc = bp[c];
    #pragma unroll
    for (int k = 0; k < DIN; k++) acc += x[n * DIN + k] * Wp[k * H + c];
    h[tid] = gelu_exact(acc);
}

// ------------------------------------------------ 2) WMMA GEMM: hp = h @ W
// One wave computes a 16x128 slab of C. W (f32, row-major [K=128][N=128]) is
// converted to f16 and staged transposed+padded in LDS so each lane's B
// fragment is 8 contiguous halves.
__global__ void __launch_bounds__(256)
gemm_nodes_wmma(const float* __restrict__ A, const float* __restrict__ W,
                float* __restrict__ C, int nrows) {
    __shared__ _Float16 Bs[H * 132];           // [n][k], padded stride 132
    int t = threadIdx.x;
    for (int idx = t; idx < H * H; idx += 256) {
        int k = idx >> 7, n = idx & 127;
        Bs[n * 132 + k] = (_Float16)W[idx];
    }
    __syncthreads();

    int wave = t >> 5, lane = t & 31;
    int m0 = (blockIdx.x * 8 + wave) * 16;     // wave-uniform guard
    if (m0 >= nrows) return;

    int mrow = m0 + (lane & 15);
    int koff = (lane >> 4) << 3;               // lanes 16-31 handle K+8
    const float* arow = A + (size_t)mrow * H;

    v8f acc[8] = {};
    for (int kb = 0; kb < H; kb += 32) {
        v16h a;
        #pragma unroll
        for (int hh = 0; hh < 8; hh++) {
            a[hh]     = (_Float16)arow[kb + koff + hh];
            a[hh + 8] = (_Float16)arow[kb + 16 + koff + hh];
        }
        #pragma unroll
        for (int nt = 0; nt < 8; nt++) {
            const _Float16* bcol = &Bs[(nt * 16 + (lane & 15)) * 132];
            v16h b;
            #pragma unroll
            for (int hh = 0; hh < 8; hh++) {
                b[hh]     = bcol[kb + koff + hh];
                b[hh + 8] = bcol[kb + 16 + koff + hh];
            }
            acc[nt] = __builtin_amdgcn_wmma_f32_16x16x32_f16(
                false, a, false, b, (short)0, acc[nt], false, false);
        }
    }
    #pragma unroll
    for (int nt = 0; nt < 8; nt++) {
        int col = nt * 16 + (lane & 15);
        #pragma unroll
        for (int r = 0; r < 8; r++) {
            int row = m0 + r + ((lane >> 4) << 3);
            C[(size_t)row * H + col] = acc[nt][r];
        }
    }
}

// ------------------------------------------------ 3) attention coefficients
__global__ void attn_coef_kernel(const float* __restrict__ hp,
                                 const float* __restrict__ a_s,
                                 const float* __restrict__ a_d,
                                 float* __restrict__ als,
                                 float* __restrict__ ald) {
    int tid = blockIdx.x * blockDim.x + threadIdx.x;
    if (tid >= N_NODES * HEADS) return;
    int n = tid >> 2, hd = tid & 3;
    const float* row = hp + (size_t)n * H + hd * DOUT;
    float s1 = 0.f, s2 = 0.f;
    #pragma unroll
    for (int d = 0; d < DOUT; d++) {
        float v = row[d];
        s1 += v * a_s[hd * DOUT + d];
        s2 += v * a_d[hd * DOUT + d];
    }
    als[tid] = s1;
    ald[tid] = s2;
}

// ------------------------------------------------ 4) edge pass 1: segment max
__global__ void edge_max_kernel(const int* __restrict__ ei,
                                const float* __restrict__ als,
                                const float* __restrict__ ald,
                                unsigned* __restrict__ menc) {
    int e = blockIdx.x * blockDim.x + threadIdx.x;
    if (e >= ETOT) return;
    int s, d; edge_endpoints(ei, e, s, d);
    #pragma unroll
    for (int hd = 0; hd < HEADS; hd++) {
        float l = als[s * HEADS + hd] + ald[d * HEADS + hd];
        l = (l > 0.f) ? l : NEG_SLOPE * l;
        atomicMax(&menc[d * HEADS + hd], fenc(l));
    }
}

// ------------------------------------------------ 5) edge pass 2: exp + denom
__global__ void edge_exp_kernel(const int* __restrict__ ei,
                                const float* __restrict__ als,
                                const float* __restrict__ ald,
                                const unsigned* __restrict__ menc,
                                float* __restrict__ evals,
                                float* __restrict__ denom) {
    int e = blockIdx.x * blockDim.x + threadIdx.x;
    if (e >= ETOT) return;
    int s, d; edge_endpoints(ei, e, s, d);
    #pragma unroll
    for (int hd = 0; hd < HEADS; hd++) {
        float l = als[s * HEADS + hd] + ald[d * HEADS + hd];
        l = (l > 0.f) ? l : NEG_SLOPE * l;
        float ev = expf(l - fdec(menc[d * HEADS + hd]));
        evals[(size_t)e * HEADS + hd] = ev;
        atomicAdd(&denom[d * HEADS + hd], ev);
    }
}

// ------------------------------------------------ 6) edge pass 3: scatter msgs
__global__ void edge_scatter_kernel(const int* __restrict__ ei,
                                    const float* __restrict__ hp,
                                    const float* __restrict__ evals,
                                    const float* __restrict__ denom,
                                    float* __restrict__ hnew) {
    long long tid = (long long)blockIdx.x * blockDim.x + threadIdx.x;
    if (tid >= (long long)ETOT * H) return;
    int e = (int)(tid >> 7), dim = (int)(tid & 127);
    int s, d; edge_endpoints(ei, e, s, d);
    int hd = dim >> 5;
    float alpha = evals[(size_t)e * HEADS + hd] /
                  (denom[d * HEADS + hd] + 1e-16f);
    atomicAdd(&hnew[(size_t)d * H + dim], hp[(size_t)s * H + dim] * alpha);
}

// ------------------------------------------------ 7) residual + LayerNorm
__global__ void __launch_bounds__(256)
residual_ln_kernel(float* __restrict__ h, const float* __restrict__ hnew,
                   const float* __restrict__ bias, const float* __restrict__ gamma,
                   const float* __restrict__ beta, int nrows) {
    int wave = threadIdx.x >> 5, lane = threadIdx.x & 31;
    int n = blockIdx.x * 8 + wave;
    if (n >= nrows) return;
    float v[4];
    #pragma unroll
    for (int j = 0; j < 4; j++) {
        int d = lane + 32 * j;
        v[j] = hnew[(size_t)n * H + d] + bias[d] + h[(size_t)n * H + d];
    }
    float s = v[0] + v[1] + v[2] + v[3];
    for (int off = 16; off; off >>= 1) s += __shfl_xor(s, off, 32);
    float mu = s * (1.0f / H);
    float q = 0.f;
    #pragma unroll
    for (int j = 0; j < 4; j++) { float t = v[j] - mu; q += t * t; }
    for (int off = 16; off; off >>= 1) q += __shfl_xor(q, off, 32);
    float inv = rsqrtf(q * (1.0f / H) + 1e-5f);
    #pragma unroll
    for (int j = 0; j < 4; j++) {
        int d = lane + 32 * j;
        h[(size_t)n * H + d] = (v[j] - mu) * inv * gamma[d] + beta[d];
    }
}

// ------------------------------------------------ 8) graph pooling
__global__ void pool_kernel(const float* __restrict__ h,
                            const int* __restrict__ n2g,
                            float* __restrict__ psum,
                            unsigned* __restrict__ pmax,
                            float* __restrict__ pcnt) {
    int tid = blockIdx.x * blockDim.x + threadIdx.x;
    if (tid >= N_NODES * H) return;
    int n = tid >> 7, d = tid & 127;
    int g = n2g[n];
    float val = h[tid];
    atomicAdd(&psum[g * H + d], val);
    atomicMax(&pmax[g * H + d], fenc(val));
    if (d == 0) atomicAdd(&pcnt[g], 1.0f);
}

__global__ void emb_kernel(const float* __restrict__ psum,
                           const unsigned* __restrict__ pmax,
                           const float* __restrict__ pcnt,
                           float* __restrict__ emb) {
    int tid = blockIdx.x * blockDim.x + threadIdx.x;
    if (tid >= NGRAPH * 2 * H) return;
    int g = tid >> 8, d = tid & 255;
    if (d < H) {
        float c = pcnt[g]; if (c < 1.0f) c = 1.0f;
        emb[tid] = psum[g * H + d] / c;
    } else {
        emb[tid] = fdec(pmax[g * H + (d - H)]);
    }
}

// ------------------------------------------------ 9) head MLP
__global__ void mlp1_kernel(const float* __restrict__ emb,
                            const float* __restrict__ Wq1,
                            const float* __restrict__ bq1,
                            float* __restrict__ p1) {
    int tid = blockIdx.x * blockDim.x + threadIdx.x;
    if (tid >= NGRAPH * H) return;
    int g = tid >> 7, c = tid & 127;
    float acc = bq1[c];
    for (int k = 0; k < 2 * H; k++) acc += emb[g * 2 * H + k] * Wq1[k * H + c];
    p1[tid] = acc;
}

__global__ void __launch_bounds__(256)
ln_gelu_kernel(float* __restrict__ p1, const float* __restrict__ gq,
               const float* __restrict__ beq) {
    int wave = threadIdx.x >> 5, lane = threadIdx.x & 31;
    int g = blockIdx.x * 8 + wave;
    if (g >= NGRAPH) return;
    float v[4];
    #pragma unroll
    for (int j = 0; j < 4; j++) v[j] = p1[g * H + lane + 32 * j];
    float s = v[0] + v[1] + v[2] + v[3];
    for (int off = 16; off; off >>= 1) s += __shfl_xor(s, off, 32);
    float mu = s * (1.0f / H);
    float q = 0.f;
    #pragma unroll
    for (int j = 0; j < 4; j++) { float t = v[j] - mu; q += t * t; }
    for (int off = 16; off; off >>= 1) q += __shfl_xor(q, off, 32);
    float inv = rsqrtf(q * (1.0f / H) + 1e-5f);
    #pragma unroll
    for (int j = 0; j < 4; j++) {
        int d = lane + 32 * j;
        p1[g * H + d] = gelu_exact((v[j] - mu) * inv * gq[d] + beq[d]);
    }
}

__global__ void mlp2_kernel(const float* __restrict__ p1,
                            const float* __restrict__ Wq2,
                            const float* __restrict__ bq2,
                            float* __restrict__ out) {
    int tid = blockIdx.x * blockDim.x + threadIdx.x;
    if (tid >= NGRAPH * FOUT) return;
    int g = tid >> 8, c = tid & 255;
    float acc = bq2[c];
    for (int k = 0; k < H; k++) acc += p1[g * H + k] * Wq2[k * FOUT + c];
    out[tid] = acc;
}

// ---------------------------------------------------------------- launch
static inline int cdiv(long long a, long long b) { return (int)((a + b - 1) / b); }

extern "C" void kernel_launch(void* const* d_in, const int* in_sizes, int n_in,
                              void* d_out, int out_size, void* d_ws, size_t ws_size,
                              hipStream_t stream) {
    const float* x   = (const float*)d_in[0];
    const int*   ei  = (const int*)d_in[1];
    const int*   n2g = (const int*)d_in[2];
    const float* Wp  = (const float*)d_in[3];
    const float* bp  = (const float*)d_in[4];
    const float* Wq1 = (const float*)d_in[5];
    const float* bq1 = (const float*)d_in[6];
    const float* gq  = (const float*)d_in[7];
    const float* beq = (const float*)d_in[8];
    const float* Wq2 = (const float*)d_in[9];
    const float* bq2 = (const float*)d_in[10];
    const float* Wg[2]  = {(const float*)d_in[11], (const float*)d_in[17]};
    const float* As[2]  = {(const float*)d_in[12], (const float*)d_in[18]};
    const float* Ad[2]  = {(const float*)d_in[13], (const float*)d_in[19]};
    const float* Bg[2]  = {(const float*)d_in[14], (const float*)d_in[20]};
    const float* Gl[2]  = {(const float*)d_in[15], (const float*)d_in[21]};
    const float* Bl[2]  = {(const float*)d_in[16], (const float*)d_in[22]};

    char* ws = (char*)d_ws;
    size_t off = 0;
    auto take = [&](size_t bytes) -> char* {
        char* p = ws + off;
        off = (off + bytes + 255) & ~(size_t)255;
        return p;
    };
    float*    h     = (float*)   take((size_t)N_NODES * H * 4);
    float*    hp    = (float*)   take((size_t)N_NODES * H * 4);
    float*    hnew  = (float*)   take((size_t)N_NODES * H * 4);
    float*    als   = (float*)   take((size_t)N_NODES * HEADS * 4);
    float*    ald   = (float*)   take((size_t)N_NODES * HEADS * 4);
    unsigned* menc  = (unsigned*)take((size_t)N_NODES * HEADS * 4);
    float*    denom = (float*)   take((size_t)N_NODES * HEADS * 4);
    float*    evals = (float*)   take((size_t)ETOT * HEADS * 4);
    float*    psum  = (float*)   take((size_t)NGRAPH * H * 4);
    unsigned* pmax  = (unsigned*)take((size_t)NGRAPH * H * 4);
    float*    pcnt  = (float*)   take((size_t)NGRAPH * 4);
    float*    emb   = (float*)   take((size_t)NGRAPH * 2 * H * 4);
    float*    p1    = (float*)   take((size_t)NGRAPH * H * 4);

    const int B = 256;

    // input projection + GELU
    proj_gelu_kernel<<<cdiv((long long)N_NODES * H, B), B, 0, stream>>>(x, Wp, bp, h);

    // two GAT layers
    for (int L = 0; L < 2; L++) {
        gemm_nodes_wmma<<<cdiv(N_NODES / 16, 8), B, 0, stream>>>(h, Wg[L], hp, N_NODES);
        attn_coef_kernel<<<cdiv((long long)N_NODES * HEADS, B), B, 0, stream>>>(
            hp, As[L], Ad[L], als, ald);

        hipMemsetAsync(menc,  0, (size_t)N_NODES * HEADS * 4, stream);
        hipMemsetAsync(denom, 0, (size_t)N_NODES * HEADS * 4, stream);
        hipMemsetAsync(hnew,  0, (size_t)N_NODES * H * 4,     stream);

        edge_max_kernel<<<cdiv(ETOT, B), B, 0, stream>>>(ei, als, ald, menc);
        edge_exp_kernel<<<cdiv(ETOT, B), B, 0, stream>>>(ei, als, ald, menc, evals, denom);
        edge_scatter_kernel<<<cdiv((long long)ETOT * H, B), B, 0, stream>>>(
            ei, hp, evals, denom, hnew);

        residual_ln_kernel<<<cdiv(N_NODES, 8), B, 0, stream>>>(
            h, hnew, Bg[L], Gl[L], Bl[L], N_NODES);
    }

    // pooling
    hipMemsetAsync(psum, 0, (size_t)NGRAPH * H * 4, stream);
    hipMemsetAsync(pmax, 0, (size_t)NGRAPH * H * 4, stream);
    hipMemsetAsync(pcnt, 0, (size_t)NGRAPH * 4,     stream);
    pool_kernel<<<cdiv((long long)N_NODES * H, B), B, 0, stream>>>(h, n2g, psum, pmax, pcnt);
    emb_kernel<<<cdiv(NGRAPH * 2 * H, B), B, 0, stream>>>(psum, pmax, pcnt, emb);

    // head MLP
    mlp1_kernel<<<cdiv(NGRAPH * H, B), B, 0, stream>>>(emb, Wq1, bq1, p1);
    ln_gelu_kernel<<<cdiv(NGRAPH, 8), B, 0, stream>>>(p1, gq, beq);
    mlp2_kernel<<<cdiv(NGRAPH * FOUT, B), B, 0, stream>>>(p1, Wq2, bq2, (float*)d_out);
}
